// TimingMeta_62440234549809
// MI455X (gfx1250) — matrix operand order repaired and checked by reading, not compiled, not run on hardware
//
#include <hip/hip_runtime.h>

// ---------------------------------------------------------------------------
// 2-layer LSTM (B=64, T=128, IN=257, H=1024) + tanh + Linear(H->2), bf16 WMMA.
// ---------------------------------------------------------------------------

#define B_  64
#define T_  128
#define IN_ 257
#define H_  1024
#define G_  4096          // 4*H
#define M_  8192          // B*T
#define KP0 288           // IN padded to multiple of 32
#define KS0 9             // KP0/32
#define KS1 32            // H/32
#define NT_ 256           // G/16 column tiles

typedef __attribute__((ext_vector_type(16))) __bf16 v16bf;
typedef __attribute__((ext_vector_type(8)))  float  v8f;

union V16A { uint4 u[2]; v16bf v; };
union V16P { __bf16 h[16]; v16bf v; };

__device__ __forceinline__ float sigmoidf_(float x) {
    return 1.0f / (1.0f + __expf(-x));
}

// ---------------------------------------------------------------------------
// Pack W [4096 x K] (f32, row-major) into WMMA-B order, K zero-padded to
// ksteps*32.  Element layout per (ntile, kstep): lane<16 -> col n0+lane,
// K = kb..kb+15 ; lane>=16 -> col n0+lane-16, K = kb+16..kb+31.  Each lane's
// 16 bf16 values are contiguous (32B) so the GEMM does one aligned v16bf load.
// ---------------------------------------------------------------------------
__global__ void pack_w_kernel(const float* __restrict__ W, __bf16* __restrict__ Bp,
                              int K, int ksteps) {
    int gid = blockIdx.x * blockDim.x + threadIdx.x;
    int total = NT_ * ksteps * 32;
    if (gid >= total) return;
    int lane = gid & 31;
    int s    = (gid >> 5) % ksteps;
    int nt   = gid / (32 * ksteps);
    int n    = nt * 16 + (lane & 15);
    int k0   = s * 32 + ((lane >> 4) ? 16 : 0);
    V16P p;
#pragma unroll
    for (int j = 0; j < 16; ++j) {
        int k = k0 + j;
        p.h[j] = (k < K) ? (__bf16)W[(long)n * K + k] : (__bf16)0.0f;
    }
    *(v16bf*)(Bp + ((long)gid << 4)) = p.v;
}

// X [8192 x 257] f32 -> Xbf [8192 x 288] bf16 (zero padded)
__global__ void conv_x_kernel(const float* __restrict__ X, __bf16* __restrict__ Xbf) {
    long gid = (long)blockIdx.x * blockDim.x + threadIdx.x;
    if (gid >= (long)M_ * KP0) return;
    long row = gid / KP0;
    int  d   = (int)(gid % KP0);
    Xbf[gid] = (d < IN_) ? (__bf16)X[row * IN_ + d] : (__bf16)0.0f;
}

// bsum0 = bih0 + bhh0 ; bsum1 = bih1 + bhh1
__global__ void bias_sum_kernel(const float* bih0, const float* bhh0,
                                const float* bih1, const float* bhh1,
                                float* bsum0, float* bsum1) {
    int gid = blockIdx.x * blockDim.x + threadIdx.x;
    if (gid < G_)            bsum0[gid]       = bih0[gid] + bhh0[gid];
    else if (gid < 2 * G_)   bsum1[gid - G_]  = bih1[gid - G_] + bhh1[gid - G_];
}

// h state (bf16) and c state (f32) from h0/c0 layer slice
__global__ void init_state_kernel(const float* __restrict__ h0, const float* __restrict__ c0,
                                  int layerOff, __bf16* __restrict__ h, float* __restrict__ c) {
    int gid = blockIdx.x * blockDim.x + threadIdx.x;
    if (gid >= B_ * H_) return;
    h[gid] = (__bf16)h0[layerOff + gid];
    c[gid] = c0[layerOff + gid];
}

// ---------------------------------------------------------------------------
// Batched GEMM with bias:  out[M x 4096] = A[M x lda] @ Bp + bias   (f32 out)
// Each wave computes FOUR consecutive 16x16 N-tiles for one 16-row M block,
// so each A fragment feeds 4 WMMAs (4x less redundant A traffic).
// 4 waves / block (one M sub-block each).  grid = (64, M/64).
// ---------------------------------------------------------------------------
__global__ __launch_bounds__(128) void gemm_bias_kernel(
        const __bf16* __restrict__ A, const __bf16* __restrict__ Bp,
        const float* __restrict__ bias, float* __restrict__ out,
        int lda, int ksteps) {
    const int lane = threadIdx.x & 31;
    const int wave = threadIdx.x >> 5;
    const int l15  = lane & 15;
    const int lhi  = lane >> 4;
    const int m0   = blockIdx.y * 64 + wave * 16;
    const int ntg  = blockIdx.x;                   // group of 4 n-tiles

    const __bf16* aBase = A + (long)(m0 + l15) * lda + lhi * 16;

    v8f acc[4] = {{}, {}, {}, {}};
    const __bf16* bBase[4];
#pragma unroll
    for (int g = 0; g < 4; ++g) {
        int nt = ntg * 4 + g;
        bBase[g] = Bp + (long)nt * ksteps * 512 + lane * 16;
    }

    for (int s = 0; s < ksteps; ++s) {
        V16A a;
        a.u[0] = *(const uint4*)(aBase + s * 32);
        a.u[1] = *(const uint4*)(aBase + s * 32 + 8);
#pragma unroll
        for (int g = 0; g < 4; ++g) {
            v16bf b = *(const v16bf*)(bBase[g] + s * 512);
            acc[g] = __builtin_amdgcn_wmma_f32_16x16x32_bf16(false, a.v, false, b,
                                                             (short)0, acc[g], false, false);
        }
    }

#pragma unroll
    for (int g = 0; g < 4; ++g) {
        const int col = (ntg * 4 + g) * 16 + l15;
        const float bs = bias[col];
#pragma unroll
        for (int r = 0; r < 8; ++r) {
            int row = m0 + r + lhi * 8;
            out[(long)row * G_ + col] = acc[g][r] + bs;
        }
    }
}

// ---------------------------------------------------------------------------
// One LSTM timestep, fused:  gates = xg[:,t,:] + h_in @ Whh^T ; cell update.
// grid = 64 blocks (one per 16-hidden-col block) x 128 threads (4 waves, one
// per 16-batch block).  Each wave computes the i,f,g,o tiles for its block;
// the C/D layout puts identical (b,h) at identical (lane,vgpr) across all 4
// accumulators, so the cell update is pure per-lane math.
// ---------------------------------------------------------------------------
__global__ __launch_bounds__(128) void lstm_step_kernel(
        const float*  __restrict__ xg,    // [8192 x 4096], row = b*T + t
        const __bf16* __restrict__ Bp,    // packed Whh (ksteps=32)
        const __bf16* __restrict__ hin,   // [64 x 1024]
        __bf16*       __restrict__ hout,  // [64 x 1024]
        float*        __restrict__ c,     // [64 x 1024]
        __bf16*       __restrict__ y,     // [8192 x 1024], row = b*T + t
        int t) {
    const int lane = threadIdx.x & 31;
    const int wave = threadIdx.x >> 5;
    const int l15  = lane & 15;
    const int lhi  = lane >> 4;
    const int m0   = wave * 16;          // batch block
    const int h0c  = blockIdx.x * 16;    // hidden column block

    const __bf16* aBase = hin + (long)(m0 + l15) * H_ + lhi * 16;

    v8f acc[4] = {{}, {}, {}, {}};
    const __bf16* bBase[4];
#pragma unroll
    for (int g = 0; g < 4; ++g) {
        int nt = g * 64 + blockIdx.x;                 // (g*1024 + h0c)/16
        bBase[g] = Bp + (long)nt * 16384 + lane * 16; // 32*32*16 elems per tile
    }

    for (int s = 0; s < 32; ++s) {
        V16A a;
        a.u[0] = *(const uint4*)(aBase + s * 32);
        a.u[1] = *(const uint4*)(aBase + s * 32 + 8);
#pragma unroll
        for (int g = 0; g < 4; ++g) {
            v16bf b = *(const v16bf*)(bBase[g] + s * 512);
            acc[g] = __builtin_amdgcn_wmma_f32_16x16x32_bf16(false, a.v, false, b,
                                                             (short)0, acc[g], false, false);
        }
    }

    const int hc = h0c + l15;
#pragma unroll
    for (int r = 0; r < 8; ++r) {
        int  b    = m0 + r + lhi * 8;
        long xrow = ((long)b * T_ + t) * G_;
        float pi = acc[0][r] + xg[xrow + 0 * H_ + hc];
        float pf = acc[1][r] + xg[xrow + 1 * H_ + hc];
        float pg = acc[2][r] + xg[xrow + 2 * H_ + hc];
        float po = acc[3][r] + xg[xrow + 3 * H_ + hc];

        float iv = sigmoidf_(pi);
        float fv = sigmoidf_(pf);
        float gv = tanhf(pg);
        float ov = sigmoidf_(po);

        long  sidx = (long)b * H_ + hc;
        float cn   = fv * c[sidx] + iv * gv;
        float hn   = ov * tanhf(cn);
        c[sidx]    = cn;
        hout[sidx] = (__bf16)hn;
        y[((long)b * T_ + t) * H_ + hc] = (__bf16)hn;
    }
}

// out[row,0:2] = tanh(y2[row,:]) . Wy[0/1,:] + by      (1 wave per row)
__global__ __launch_bounds__(256) void final_proj_kernel(
        const __bf16* __restrict__ y2, const float* __restrict__ Wy,
        const float* __restrict__ by, float* __restrict__ out) {
    int gid  = blockIdx.x * blockDim.x + threadIdx.x;
    int row  = gid >> 5;
    int lane = gid & 31;
    if (row >= M_) return;
    const __bf16* yr = y2 + (long)row * H_;
    float s0 = 0.0f, s1 = 0.0f;
    for (int j = lane; j < H_; j += 32) {
        float v = tanhf((float)yr[j]);
        s0 += v * Wy[j];
        s1 += v * Wy[H_ + j];
    }
#pragma unroll
    for (int off = 16; off > 0; off >>= 1) {
        s0 += __shfl_down(s0, off, 32);
        s1 += __shfl_down(s1, off, 32);
    }
    if (lane == 0) {
        out[(long)row * 2 + 0] = s0 + by[0];
        out[(long)row * 2 + 1] = s1 + by[1];
    }
}

// ---------------------------------------------------------------------------
extern "C" void kernel_launch(void* const* d_in, const int* in_sizes, int n_in,
                              void* d_out, int out_size, void* d_ws, size_t ws_size,
                              hipStream_t stream) {
    (void)in_sizes; (void)n_in; (void)out_size; (void)ws_size;

    const float* X    = (const float*)d_in[0];
    const float* h0   = (const float*)d_in[1];
    const float* c0   = (const float*)d_in[2];
    const float* Wih0 = (const float*)d_in[3];
    const float* Whh0 = (const float*)d_in[4];
    const float* bih0 = (const float*)d_in[5];
    const float* bhh0 = (const float*)d_in[6];
    const float* Wih1 = (const float*)d_in[7];
    const float* Whh1 = (const float*)d_in[8];
    const float* bih1 = (const float*)d_in[9];
    const float* bhh1 = (const float*)d_in[10];
    const float* Wy   = (const float*)d_in[11];
    const float* by   = (const float*)d_in[12];
    float* out = (float*)d_out;

    // ---- workspace layout (256B aligned slices) ----
    char*  ws  = (char*)d_ws;
    size_t off = 0;
    auto take = [&](size_t bytes) -> char* {
        char* p = ws + off;
        off += (bytes + 255) & ~(size_t)255;
        return p;
    };
    __bf16* Wp0  = (__bf16*)take((size_t)NT_ * KS0 * 32 * 16 * 2); // packed Wih0 (K padded 288)
    __bf16* Wh0  = (__bf16*)take((size_t)NT_ * KS1 * 32 * 16 * 2); // packed Whh0
    __bf16* Wi1  = (__bf16*)take((size_t)NT_ * KS1 * 32 * 16 * 2); // packed Wih1
    __bf16* Wh1  = (__bf16*)take((size_t)NT_ * KS1 * 32 * 16 * 2); // packed Whh1
    __bf16* Xbf  = (__bf16*)take((size_t)M_ * KP0 * 2);            // padded bf16 X
    float*  xg   = (float*) take((size_t)M_ * G_ * 4);             // gate pre-acts (reused L0,L1)
    __bf16* ybuf = (__bf16*)take((size_t)M_ * H_ * 2);             // y1, then reused as y2
    __bf16* hA   = (__bf16*)take((size_t)B_ * H_ * 2);
    __bf16* hB   = (__bf16*)take((size_t)B_ * H_ * 2);
    float*  cbuf = (float*) take((size_t)B_ * H_ * 4);
    float*  bs0  = (float*) take((size_t)G_ * 4);
    float*  bs1  = (float*) take((size_t)G_ * 4);

    // ---- preprocessing ----
    pack_w_kernel<<<(NT_ * KS0 * 32 + 255) / 256, 256, 0, stream>>>(Wih0, Wp0, IN_, KS0);
    pack_w_kernel<<<(NT_ * KS1 * 32 + 255) / 256, 256, 0, stream>>>(Whh0, Wh0, H_, KS1);
    pack_w_kernel<<<(NT_ * KS1 * 32 + 255) / 256, 256, 0, stream>>>(Wih1, Wi1, H_, KS1);
    pack_w_kernel<<<(NT_ * KS1 * 32 + 255) / 256, 256, 0, stream>>>(Whh1, Wh1, H_, KS1);
    conv_x_kernel<<<(int)(((long)M_ * KP0 + 255) / 256), 256, 0, stream>>>(X, Xbf);
    bias_sum_kernel<<<(2 * G_ + 255) / 256, 256, 0, stream>>>(bih0, bhh0, bih1, bhh1, bs0, bs1);

    // ---- layer 0 ----
    init_state_kernel<<<(B_ * H_ + 255) / 256, 256, 0, stream>>>(h0, c0, 0, hA, cbuf);
    gemm_bias_kernel<<<dim3(NT_ / 4, M_ / 64), 128, 0, stream>>>(Xbf, Wp0, bs0, xg, KP0, KS0);
    for (int t = 0; t < T_; ++t) {
        const __bf16* hin  = (t & 1) ? hB : hA;
        __bf16*       hout = (t & 1) ? hA : hB;
        lstm_step_kernel<<<H_ / 16, 128, 0, stream>>>(xg, Wh0, hin, hout, cbuf, ybuf, t);
    }

    // ---- layer 1 ----
    gemm_bias_kernel<<<dim3(NT_ / 4, M_ / 64), 128, 0, stream>>>(ybuf, Wi1, bs1, xg, H_, KS1);
    init_state_kernel<<<(B_ * H_ + 255) / 256, 256, 0, stream>>>(h0, c0, B_ * H_, hA, cbuf);
    for (int t = 0; t < T_; ++t) {
        const __bf16* hin  = (t & 1) ? hB : hA;
        __bf16*       hout = (t & 1) ? hA : hB;
        lstm_step_kernel<<<H_ / 16, 128, 0, stream>>>(xg, Wh1, hin, hout, cbuf, ybuf, t);
    }

    // ---- head ----
    final_proj_kernel<<<(M_ * 32) / 256, 256, 0, stream>>>(ybuf, Wy, by, out);
}
